// RegionProposalNetwork_22935125361088
// MI455X (gfx1250) — compile-verified, matrix operand order checked by reading, not run on hardware
//
#include <hip/hip_runtime.h>
#include <hip/hip_bf16.h>
#include <math.h>

typedef _Float16 half8  __attribute__((ext_vector_type(8)));
typedef _Float16 half16 __attribute__((ext_vector_type(16)));
typedef float    float8 __attribute__((ext_vector_type(8)));

#define HH 128
#define WW 128
#define CIN 512
#define NPIX (HH * WW)            // 16384
#define KDIM (9 * CIN)            // 4608
#define NANC 9
#define NBOX (NPIX * NANC)        // 147456
#define NPOST 300

// d_out float offsets
#define LOC_OFF    0
#define SCORE_OFF  589824
#define ROI_OFF    884736
#define ANCH_OFF   885936

__device__ __forceinline__ half16 ld_frag(const _Float16* p) {
    half8 lo = *(const half8*)p;
    half8 hi = *(const half8*)(p + 16);
    return __builtin_shufflevector(lo, hi, 0,1,2,3,4,5,6,7,8,9,10,11,12,13,14,15);
}

// ---------------------------------------------------------------- pack kernels
__global__ void pack_x_f16(const float* __restrict__ x, _Float16* __restrict__ xh,
                           _Float16* __restrict__ zpad, int n) {
    int tid0 = blockIdx.x * blockDim.x + threadIdx.x;
    for (int i = tid0; i < n; i += gridDim.x * blockDim.x)
        xh[i] = (_Float16)x[i];
    if (tid0 < 512) zpad[tid0] = (_Float16)0.f;       // zero page for OOB taps
}

// conv_w HWIO (ky,kx,cin,cout) -> wt[cout][K], K=(ky*3+kx)*512+cin
__global__ void pack_w_f16(const float* __restrict__ w, _Float16* __restrict__ wt) {
    int i = blockIdx.x * blockDim.x + threadIdx.x;   // i = cout*KDIM + K
    if (i >= 512 * KDIM) return;
    int cout = i / KDIM;
    int K    = i - cout * KDIM;
    wt[i] = (_Float16)w[K * 512 + cout];
}

// pack loc_w (512x36) + score_w (512x18) -> wt2[col][cin], col 0..63 (54..63 zero)
__global__ void pack_w2_f16(const float* __restrict__ loc_w, const float* __restrict__ score_w,
                            _Float16* __restrict__ wt2) {
    int i = blockIdx.x * blockDim.x + threadIdx.x;   // col*512 + cin
    if (i >= 64 * 512) return;
    int col = i >> 9;
    int cin = i & 511;
    float v = 0.f;
    if (col < 36)      v = loc_w[cin * 36 + col];
    else if (col < 54) v = score_w[cin * 18 + (col - 36)];
    wt2[i] = (_Float16)v;
}

// ------------------------------------------------------- 3x3 conv implicit GEMM
// M=16384 (pixels), N=512, K=4608.  One wave -> 64x64 C supertile (4x4 WMMA tiles).
// 256 M-supertiles x 8 N-supertiles = 2048 waves = 256 blocks x 8 waves.
__global__ void conv3x3_wmma(const _Float16* __restrict__ xh, const _Float16* __restrict__ wt,
                             const _Float16* __restrict__ zpad,
                             const float* __restrict__ conv_b, _Float16* __restrict__ h16) {
    const int wave = threadIdx.x >> 5;
    const int lane = threadIdx.x & 31;
    const int t  = blockIdx.x * 8 + wave;             // 0..2047
    const int sN = t & 7;                             // 8 N supertiles
    const int sM = t >> 3;                            // 256 M supertiles
    const int m0 = sM << 6, n0 = sN << 6;
    const int lrow = lane & 15;
    const int kb   = (lane < 16) ? 0 : 8;
    const int rbase = (lane >= 16) ? 8 : 0;

    float8 acc[4][4] = {};

#pragma unroll
    for (int tap = 0; tap < 9; ++tap) {
        const int dy = tap / 3 - 1, dx = tap % 3 - 1; // folds to constants
        // per-tap A base pointers (zero page when out of bounds) -> branch-free inner loop
        const _Float16* ap[4];
#pragma unroll
        for (int i = 0; i < 4; ++i) {
            int m = m0 + (i << 4) + lrow;
            int py = m >> 7, px = m & 127;
            int ny = py + dy, nx = px + dx;
            bool inb = ((unsigned)ny < 128u) && ((unsigned)nx < 128u);
            ap[i] = (inb ? (xh + (((ny << 7) + nx) << 9)) : zpad) + kb;
        }
        const _Float16* bp = wt + (n0 + lrow) * KDIM + (tap << 9) + kb;

        for (int c0 = 0; c0 < 512; c0 += 32) {
            half16 A[4];
#pragma unroll
            for (int i = 0; i < 4; ++i) A[i] = ld_frag(ap[i] + c0);
#pragma unroll
            for (int j = 0; j < 4; ++j) {
                half16 B = ld_frag(bp + ((j * 16) * KDIM) + c0);
#pragma unroll
                for (int i = 0; i < 4; ++i)
                    acc[i][j] = __builtin_amdgcn_wmma_f32_16x16x32_f16(
                        false, A[i], false, B, (short)0, acc[i][j], false, false);
            }
        }
    }

#pragma unroll
    for (int j = 0; j < 4; ++j) {
        const int col = n0 + (j << 4) + lrow;
        const float bias = conv_b[col];
#pragma unroll
        for (int i = 0; i < 4; ++i) {
#pragma unroll
            for (int r = 0; r < 8; ++r) {
                int row = m0 + (i << 4) + r + rbase;
                float v = acc[i][j][r] + bias;
                v = v > 0.f ? v : 0.f;                // ReLU
                h16[(row << 9) + col] = (_Float16)v;
            }
        }
    }
}

// ------------------------------------------------------- 1x1 convs (loc+score)
// M=16384, N=64 (36 loc + 18 score + 10 pad), K=512.
__global__ void conv1x1_wmma(const _Float16* __restrict__ h16, const _Float16* __restrict__ wt2,
                             const float* __restrict__ loc_b, const float* __restrict__ score_b,
                             float* __restrict__ out) {
    const int wave = threadIdx.x >> 5;
    const int lane = threadIdx.x & 31;
    const int t     = blockIdx.x * 8 + wave;          // 0..4095
    const int tileN = t & 3;                          // 4 N-tiles
    const int tileM = t >> 2;                         // 1024 M-tiles
    const int m0 = tileM << 4, n0 = tileN << 4;
    const int lrow = lane & 15;
    const int kb   = (lane < 16) ? 0 : 8;
    const int col  = n0 + lrow;

    const _Float16* ap = h16 + ((m0 + lrow) << 9) + kb;
    const _Float16* bp = wt2 + (col << 9) + kb;

    float8 acc = {};
    for (int k0 = 0; k0 < 512; k0 += 32) {
        half16 a = ld_frag(ap + k0);
        half16 b = ld_frag(bp + k0);
        acc = __builtin_amdgcn_wmma_f32_16x16x32_f16(false, a, false, b, (short)0, acc, false, false);
    }

    const int rbase = (lane >= 16) ? 8 : 0;
#pragma unroll
    for (int r = 0; r < 8; ++r) {
        int row = m0 + r + rbase;                     // pixel
        float v = acc[r];
        if (col < 36) {
            out[LOC_OFF + row * 36 + col] = v + loc_b[col];
        } else if (col < 54) {
            int sc = col - 36;
            out[SCORE_OFF + row * 18 + sc] = v + score_b[sc];
        }
    }
}

// ------------------------------------------- anchors, loc2bbox, clip, fg scores
__global__ void proposal_prep(const float* __restrict__ dout, float* __restrict__ anch_out,
                              float* __restrict__ boxes, float* __restrict__ scores,
                              const int* img_h_p, const int* img_w_p, const int* scale_p) {
    int i = blockIdx.x * blockDim.x + threadIdx.x;
    if (i >= NBOX) return;
    const float img_h = (float)img_h_p[0];
    const float img_w = (float)img_w_p[0];
    const float min_size = 16.f * (float)scale_p[0];

    const int p = i / NANC, a = i - p * NANC;
    const int py = p >> 7, px = p & 127;

    const float ratios[3] = {0.5f, 1.0f, 2.0f};
    const float scales[3] = {8.f, 16.f, 32.f};
    float r = ratios[a / 3], s = scales[a % 3];
    float sr = sqrtf(r);
    float ha = 16.f * s * sr;
    float wa = 16.f * s / sr;
    float cy0 = 8.f + (float)py * 16.f;
    float cx0 = 8.f + (float)px * 16.f;
    float ay1 = cy0 - 0.5f * ha, ax1 = cx0 - 0.5f * wa;
    float ay2 = cy0 + 0.5f * ha, ax2 = cx0 + 0.5f * wa;
    anch_out[i * 4 + 0] = ay1; anch_out[i * 4 + 1] = ax1;
    anch_out[i * 4 + 2] = ay2; anch_out[i * 4 + 3] = ax2;

    // loc2bbox
    const float* loc = dout + LOC_OFF + (size_t)p * 36 + a * 4;
    float sh = ay2 - ay1, sw = ax2 - ax1;
    float scy = ay1 + 0.5f * sh, scx = ax1 + 0.5f * sw;
    float cy = loc[0] * sh + scy;
    float cx = loc[1] * sw + scx;
    float hh = expf(loc[2]) * sh;
    float ww = expf(loc[3]) * sw;
    float y1 = fminf(fmaxf(cy - 0.5f * hh, 0.f), img_h);
    float x1 = fminf(fmaxf(cx - 0.5f * ww, 0.f), img_w);
    float y2 = fminf(fmaxf(cy + 0.5f * hh, 0.f), img_h);
    float x2 = fminf(fmaxf(cx + 0.5f * ww, 0.f), img_w);
    boxes[i * 4 + 0] = y1; boxes[i * 4 + 1] = x1;
    boxes[i * 4 + 2] = y2; boxes[i * 4 + 3] = x2;

    bool keep = ((y2 - y1) >= min_size) && ((x2 - x1) >= min_size);
    float s0 = dout[SCORE_OFF + (size_t)p * 18 + a * 2 + 0];
    float s1 = dout[SCORE_OFF + (size_t)p * 18 + a * 2 + 1];
    float fg = 1.f / (1.f + expf(s0 - s1));           // softmax fg channel
    scores[i] = keep ? fg : -INFINITY;
}

// --------------------------------------------- sequential greedy NMS, 1 block
__global__ __launch_bounds__(1024)
void nms_kernel(const float* __restrict__ boxes, float* __restrict__ scores,
                float* __restrict__ roi) {
    __shared__ float smax[1024];
    __shared__ int   sidx[1024];
    __shared__ float cb[5];
    const int tid = threadIdx.x;

    for (int it = 0; it < NPOST; ++it) {
        float best = -INFINITY; int bi = NBOX;        // sentinel
        for (int j = tid; j < NBOX; j += 1024) {
            float sc = scores[j];
            if (sc > best) { best = sc; bi = j; }     // strict > : lowest index wins
        }
        smax[tid] = best; sidx[tid] = bi;
        __syncthreads();
        for (int off = 512; off > 0; off >>= 1) {
            if (tid < off) {
                float so = smax[tid + off]; int io = sidx[tid + off];
                if (so > smax[tid] || (so == smax[tid] && io < sidx[tid])) {
                    smax[tid] = so; sidx[tid] = io;
                }
            }
            __syncthreads();
        }
        if (tid == 0) {
            int i = sidx[0]; float sc = smax[0];
            bool ok = (sc > -1e30f) && (i < NBOX);
            float y1 = 0.f, x1 = 0.f, y2 = 0.f, x2 = 0.f;
            if (i < NBOX) {
                y1 = boxes[i * 4 + 0]; x1 = boxes[i * 4 + 1];
                y2 = boxes[i * 4 + 2]; x2 = boxes[i * 4 + 3];
            }
            float m = ok ? 1.f : 0.f;
            roi[it * 4 + 0] = y1 * m; roi[it * 4 + 1] = x1 * m;
            roi[it * 4 + 2] = y2 * m; roi[it * 4 + 3] = x2 * m;
            cb[0] = y1; cb[1] = x1; cb[2] = y2; cb[3] = x2;
            cb[4] = (y2 - y1) * (x2 - x1);
        }
        __syncthreads();
        const float cy1 = cb[0], cx1 = cb[1], cy2 = cb[2], cx2 = cb[3], ca = cb[4];
        for (int j = tid; j < NBOX; j += 1024) {
            float y1 = boxes[j * 4 + 0], x1 = boxes[j * 4 + 1];
            float y2 = boxes[j * 4 + 2], x2 = boxes[j * 4 + 3];
            float iy1 = fmaxf(y1, cy1), ix1 = fmaxf(x1, cx1);
            float iy2 = fminf(y2, cy2), ix2 = fminf(x2, cx2);
            float inter = fmaxf(iy2 - iy1, 0.f) * fmaxf(ix2 - ix1, 0.f);
            float area  = (y2 - y1) * (x2 - x1);
            float iou   = inter / (area + ca - inter + 1e-9f);
            if (iou > 0.7f) scores[j] = -INFINITY;    // suppress
        }
        __syncthreads();
    }
}

// ------------------------------------------------------------------- launcher
extern "C" void kernel_launch(void* const* d_in, const int* in_sizes, int n_in,
                              void* d_out, int out_size, void* d_ws, size_t ws_size,
                              hipStream_t stream) {
    const float* x       = (const float*)d_in[0];
    const float* conv_w  = (const float*)d_in[1];
    const float* conv_b  = (const float*)d_in[2];
    const float* loc_w   = (const float*)d_in[3];
    const float* loc_b   = (const float*)d_in[4];
    const float* score_w = (const float*)d_in[5];
    const float* score_b = (const float*)d_in[6];
    const int*   img_h   = (const int*)d_in[7];
    const int*   img_w   = (const int*)d_in[8];
    const int*   scale   = (const int*)d_in[9];
    float* out = (float*)d_out;

    // workspace carve-up (bytes)
    char* ws = (char*)d_ws;
    _Float16* xh     = (_Float16*)(ws);                      // 16 MB
    _Float16* wt     = (_Float16*)(ws + 16777216);           // 4.5 MB
    _Float16* h16    = (_Float16*)(ws + 21495808);           // 16 MB
    _Float16* wt2    = (_Float16*)(ws + 38273024);           // 64 KB
    float*    boxes  = (float*)(ws + 38338560);              // 2.25 MB
    float*    scores = (float*)(ws + 40697856);              // 576 KB
    _Float16* zpad   = (_Float16*)(ws + 41287680);           // 1 KB zero page

    // 1. pack to f16 (+ zero page)
    pack_x_f16<<<4096, 256, 0, stream>>>(x, xh, zpad, NPIX * CIN);
    pack_w_f16<<<(512 * KDIM + 255) / 256, 256, 0, stream>>>(conv_w, wt);
    pack_w2_f16<<<(64 * 512 + 255) / 256, 256, 0, stream>>>(loc_w, score_w, wt2);

    // 2. 3x3 conv + ReLU via WMMA (2048 waves, 64x64 C tile each)
    conv3x3_wmma<<<256, 256, 0, stream>>>(xh, wt, zpad, conv_b, h16);

    // 3. fused 1x1 loc+score conv via WMMA (4096 tiles)
    conv1x1_wmma<<<512, 256, 0, stream>>>(h16, wt2, loc_b, score_b, out);

    // 4. anchors + loc2bbox + clip + min-size + fg scores
    proposal_prep<<<(NBOX + 255) / 256, 256, 0, stream>>>(
        out, out + ANCH_OFF, boxes, scores, img_h, img_w, scale);

    // 5. greedy NMS -> roi (300 boxes)
    nms_kernel<<<1, 1024, 0, stream>>>(boxes, scores, out + ROI_OFF);
}
